// MMDoubleStreamBlock_8873402434116
// MI455X (gfx1250) — matrix operand order repaired and checked by reading, not compiled, not run on hardware
//
#include <hip/hip_runtime.h>
#include <math.h>

// ---------------------------------------------------------------------------
// CDNA5 / gfx1250 MMDoubleStreamBlock:
//   bf16 WMMA GEMMs fed by the Tensor Data Mover (double-buffered LDS),
//   WMMA flash attention, packed fragment-major B weights.
// ---------------------------------------------------------------------------

typedef __attribute__((ext_vector_type(16))) __bf16 v16bf;
typedef __attribute__((ext_vector_type(8)))  float  v8f;
typedef __attribute__((ext_vector_type(4)))  unsigned v4u;
typedef __attribute__((ext_vector_type(8)))  int    v8i;
typedef __attribute__((ext_vector_type(4)))  int    v4i;

union FragBF { v16bf v; __bf16 e[16]; uint4 q[2]; };
union AccF   { v8f   v; float  e[8]; };

__device__ __forceinline__ __bf16 f2bf(float f) {
  unsigned u = __builtin_bit_cast(unsigned, f);
  unsigned r = (u + 0x7FFFu + ((u >> 16) & 1u)) >> 16;
  unsigned short s = (unsigned short)r;
  return __builtin_bit_cast(__bf16, s);
}

__device__ __forceinline__ v8f wmma_bf16(v16bf a, v16bf b, v8f c) {
  return __builtin_amdgcn_wmma_f32_16x16x32_bf16(
      /*neg_a=*/false, a, /*neg_b=*/false, b,
      /*c_mod=*/(short)0, c, /*reuse_a=*/false, /*reuse_b=*/false);
}

// TDM 2D tile load global->LDS (bf16 elements, data_size = 2 bytes).
//   dim0/dim1  : tensor extents (OOB clamp), data_size units
//   tile0/tile1: tile extents, data_size units (tile0 contiguous)
//   stride0    : elements between consecutive dim-1 lines
__device__ __forceinline__ void tdm_load_2d(unsigned lds_addr, const void* gptr,
                                            unsigned dim0, unsigned dim1,
                                            unsigned tile0, unsigned tile1,
                                            unsigned long long stride0) {
  unsigned long long ga = (unsigned long long)(size_t)gptr;
  v4u g0;
  g0[0] = 1u;                                               // count=1, user D#
  g0[1] = lds_addr;                                         // LDS byte address
  g0[2] = (unsigned)(ga & 0xFFFFFFFFull);                   // global_addr lo
  g0[3] = (unsigned)((ga >> 32) & 0x1FFFFFFull) | (2u << 30); // hi | type=2
  v8i g1;
  g1[0] = (int)(1u << 16);                                  // data_size=2B
  g1[1] = (int)((dim0 & 0xFFFFu) << 16);                    // tensor_dim0 lo
  g1[2] = (int)(((dim0 >> 16) & 0xFFFFu) | ((dim1 & 0xFFFFu) << 16));
  g1[3] = (int)(((dim1 >> 16) & 0xFFFFu) | ((tile0 & 0xFFFFu) << 16));
  g1[4] = (int)(tile1 & 0xFFFFu);                           // tile_dim1, tile_dim2=0
  g1[5] = (int)(unsigned)(stride0 & 0xFFFFFFFFull);         // dim0_stride lo
  g1[6] = (int)(unsigned)((stride0 >> 32) & 0xFFFFull);     // dim0_stride hi
  g1[7] = 0;
  v4i z4; z4[0] = 0; z4[1] = 0; z4[2] = 0; z4[3] = 0;
  v8i z8; z8[0] = 0; z8[1] = 0; z8[2] = 0; z8[3] = 0;
  z8[4] = 0; z8[5] = 0; z8[6] = 0; z8[7] = 0;
  // 6-arg toolchain variant: (g0, g1, g2, g3, extra, cpol)
  __builtin_amdgcn_tensor_load_to_lds(g0, g1, z4, z4, z8, 0);
}

// ---------------------------------------------------------------------------
// Elementwise helpers
// ---------------------------------------------------------------------------
__global__ void copy_f32_kernel(const float* __restrict__ src,
                                float* __restrict__ dst, int n) {
  int i = blockIdx.x * blockDim.x + threadIdx.x;
  if (i < n) dst[i] = src[i];
}

__global__ void silu_kernel(const float* __restrict__ x,
                            float* __restrict__ y, int n) {
  int i = blockIdx.x * blockDim.x + threadIdx.x;
  if (i < n) { float v = x[i]; y[i] = v / (1.0f + expf(-v)); }
}

// Pack f32 weight W[K][N] into bf16 WMMA-B-fragment-major layout:
//   Bp[((kc*pn + nt)*16 + col)*32 + e]  = W[kc*32 + e][nt*16 + col]
__global__ void pack_b_kernel(const float* __restrict__ W,
                              __bf16* __restrict__ Bp, int Kdim, int Ndim) {
  int i = blockIdx.x * blockDim.x + threadIdx.x;
  if (i >= Kdim * Ndim) return;
  int pn   = Ndim >> 4;
  int col  = i & 15;
  int e    = (i >> 4) & 31;
  int rest = i >> 9;
  int nt   = rest % pn;
  int kc   = rest / pn;
  size_t dst = ((size_t)(kc * pn + nt) * 16 + col) * 32 + e;
  Bp[dst] = f2bf(W[(size_t)(kc * 32 + e) * Ndim + nt * 16 + col]);
}

// out[j] = sum_k vs[k] * W[k*N + j] + b[j]   (modulation GEMV)
__global__ void modvec_kernel(const float* __restrict__ vs,
                              const float* __restrict__ W,
                              const float* __restrict__ b,
                              float* __restrict__ out, int K, int N) {
  int j = blockIdx.x * blockDim.x + threadIdx.x;
  if (j >= N) return;
  float acc = 0.f;
  for (int k = 0; k < K; ++k) acc += vs[k] * W[(size_t)k * N + j];
  out[j] = acc + b[j];
}

// per-token LayerNorm + modulate + cast to bf16.  One block per token.
__global__ __launch_bounds__(256)
void ln_modulate_cast_kernel(const float* __restrict__ x,
                             const float* __restrict__ shift,
                             const float* __restrict__ scale,
                             __bf16* __restrict__ out, int D) {
  __shared__ float s_sum[256];
  __shared__ float s_sq[256];
  int t = blockIdx.x;
  const float* xr = x + (size_t)t * D;
  float sum = 0.f, sq = 0.f;
  for (int j = threadIdx.x; j < D; j += 256) {
    float v = xr[j]; sum += v; sq += v * v;
  }
  s_sum[threadIdx.x] = sum; s_sq[threadIdx.x] = sq;
  __syncthreads();
  for (int s = 128; s > 0; s >>= 1) {
    if (threadIdx.x < s) {
      s_sum[threadIdx.x] += s_sum[threadIdx.x + s];
      s_sq[threadIdx.x]  += s_sq[threadIdx.x + s];
    }
    __syncthreads();
  }
  float mean = s_sum[0] / (float)D;
  float var  = s_sq[0] / (float)D - mean * mean;
  float inv  = rsqrtf(var + 1e-6f);
  __bf16* orow = out + (size_t)t * D;
  for (int j = threadIdx.x; j < D; j += 256) {
    float y = (xr[j] - mean) * inv;
    orow[j] = f2bf(y * (1.0f + scale[j]) + shift[j]);
  }
}

// x += p * g[j]
__global__ void gate_residual_kernel(float* __restrict__ x,
                                     const float* __restrict__ p,
                                     const float* __restrict__ g,
                                     int T, int D) {
  int i = blockIdx.x * blockDim.x + threadIdx.x;
  if (i < T * D) x[i] += p[i] * g[i % D];
}

// hb = gelu_tanh(y + b[col]) as bf16
__global__ void bias_gelu_cast_kernel(const float* __restrict__ y,
                                      const float* __restrict__ b,
                                      __bf16* __restrict__ out,
                                      int total, int Ncols) {
  int i = blockIdx.x * blockDim.x + threadIdx.x;
  if (i >= total) return;
  float v = y[i] + b[i % Ncols];
  float t = tanhf(0.7978845608028654f * (v + 0.044715f * v * v * v));
  out[i] = f2bf(0.5f * v * (1.0f + t));
}

// x += (y + b[j]) * g[j]
__global__ void bias_gate_residual_kernel(float* __restrict__ x,
                                          const float* __restrict__ y,
                                          const float* __restrict__ b,
                                          const float* __restrict__ g,
                                          int T, int D) {
  int i = blockIdx.x * blockDim.x + threadIdx.x;
  if (i < T * D) { int j = i % D; x[i] += (y[i] + b[j]) * g[j]; }
}

// ---------------------------------------------------------------------------
// QKV post-process: per-head RMSNorm + (optional) RoPE, f32 -> bf16.
// Q,K row-major [L,NH,HD]; V transposed per head [NH*HD][L] for flash P@V.
// ---------------------------------------------------------------------------
__global__ __launch_bounds__(128)
void qkv_post_kernel(const float* __restrict__ qkv,      // [T, 3*NH*128]
                     const float* __restrict__ qw,
                     const float* __restrict__ kw,
                     const float* __restrict__ cosb,     // [T,128]
                     const float* __restrict__ sinb,
                     int use_rope,
                     __bf16* __restrict__ Qb,            // [L,NH,128]
                     __bf16* __restrict__ Kb,            // [L,NH,128]
                     __bf16* __restrict__ Vt,            // [NH*128][L]
                     int NH, int Ltot, int row_off) {
  __shared__ float red[128];
  const int HD = 128;
  int t = blockIdx.x, h = blockIdx.y, j = threadIdx.x;
  size_t base = (size_t)t * (3 * NH * HD) + (size_t)h * HD;
  float q  = qkv[base + j];
  float k  = qkv[base + NH * HD + j];
  float v  = qkv[base + 2 * NH * HD + j];
  float qp = qkv[base + (j ^ 1)];
  float kp = qkv[base + NH * HD + (j ^ 1)];

  red[j] = q * q; __syncthreads();
  for (int s = 64; s > 0; s >>= 1) { if (j < s) red[j] += red[j + s]; __syncthreads(); }
  float rsq = rsqrtf(red[0] * (1.0f / 128.0f) + 1e-6f);
  __syncthreads();
  red[j] = k * k; __syncthreads();
  for (int s = 64; s > 0; s >>= 1) { if (j < s) red[j] += red[j + s]; __syncthreads(); }
  float rsk = rsqrtf(red[0] * (1.0f / 128.0f) + 1e-6f);

  float qn  = q  * rsq * qw[j];
  float kn  = k  * rsk * kw[j];
  float qnp = qp * rsq * qw[j ^ 1];
  float knp = kp * rsk * kw[j ^ 1];

  if (use_rope) {
    float c = cosb[(size_t)t * HD + j];
    float s = sinb[(size_t)t * HD + j];
    float rq = (j & 1) ? qnp : -qnp;   // rotate_half
    float rk = (j & 1) ? knp : -knp;
    qn = qn * c + rq * s;
    kn = kn * c + rk * s;
  }
  size_t o = ((size_t)(row_off + t) * NH + h) * HD + j;
  Qb[o] = f2bf(qn); Kb[o] = f2bf(kn);
  Vt[((size_t)h * HD + j) * Ltot + row_off + t] = f2bf(v);
}

// ---------------------------------------------------------------------------
// Tiled bf16 WMMA GEMM fed by the Tensor Data Mover.
// C[M,N] f32 = A[M,K] bf16 @ packed-B bf16.
// Block 256 threads (8 waves), tile 128x128x32, double-buffered LDS;
// wave 0 drives tensor_load_to_lds for the next k-step while all waves WMMA.
// ---------------------------------------------------------------------------
#define GBM 128
#define GBN 128
#define GBK 32

__global__ __launch_bounds__(256)
void gemm_bf16_kernel(const __bf16* __restrict__ A,
                      const __bf16* __restrict__ Bp,    // packed layout
                      float* __restrict__ C, int Mdim, int Ndim, int Kdim) {
  __shared__ __bf16 As[2][GBM][GBK];     // 2 x 8KB
  __shared__ __bf16 Bs[2][8 * 512];      // 2 x 8KB, fragment-major panel

  int tid  = threadIdx.x;
  int lane = tid & 31;
  int lr   = lane & 15;
  int hi   = lane >> 4;
  int wid  = tid >> 5;
  int wm   = wid & 3;         // 0..3 -> 32-row strip
  int wn   = wid >> 2;        // 0..1 -> 64-col strip
  int rowBase = blockIdx.y * GBM;
  int colBase = blockIdx.x * GBN;
  int pn      = Ndim >> 4;
  int ntBase  = colBase >> 4;

  AccF acc[2][4];
  for (int a = 0; a < 2; ++a)
    for (int b = 0; b < 4; ++b)
      for (int r = 0; r < 8; ++r) acc[a][b].e[r] = 0.f;

  // scalar-branch guard: exactly one wave issues TDM ops (TDM ignores EXEC)
  bool issuer = (__builtin_amdgcn_readfirstlane(wid) == 0);

  unsigned ldsA[2] = { (unsigned)(size_t)&As[0][0][0], (unsigned)(size_t)&As[1][0][0] };
  unsigned ldsB[2] = { (unsigned)(size_t)&Bs[0][0],    (unsigned)(size_t)&Bs[1][0] };

  int nkc = Kdim >> 5;

  if (issuer) {
    // prologue: DMA k-step 0 into buffer 0
    tdm_load_2d(ldsA[0], A + (size_t)rowBase * Kdim,
                GBK, GBM, GBK, GBM, (unsigned long long)Kdim);
    tdm_load_2d(ldsB[0], Bp + (size_t)ntBase * 512,
                4096u, 1u, 4096u, 1u, 4096ull);
  }

  for (int kc = 0; kc < nkc; ++kc) {
    int cb = kc & 1;
    if (issuer) __builtin_amdgcn_s_wait_tensorcnt(0);  // buffer cb DMA done
    __syncthreads();                                   // visible to all waves
    if (issuer && (kc + 1 < nkc)) {
      // prefetch next k-step into the other buffer while we compute
      tdm_load_2d(ldsA[cb ^ 1], A + (size_t)rowBase * Kdim + (kc + 1) * GBK,
                  GBK, GBM, GBK, GBM, (unsigned long long)Kdim);
      tdm_load_2d(ldsB[cb ^ 1], Bp + ((size_t)(kc + 1) * pn + ntBase) * 512,
                  4096u, 1u, 4096u, 1u, 4096ull);
    }

    FragBF af[2], bf[4];
    for (int tm = 0; tm < 2; ++tm) {
      int r = wm * 32 + tm * 16 + lr;
      af[tm].q[0] = *(const uint4*)&As[cb][r][hi * 16];
      af[tm].q[1] = *(const uint4*)&As[cb][r][hi * 16 + 8];
    }
    for (int tn = 0; tn < 4; ++tn) {
      int t_loc = wn * 4 + tn;
      const uint4* p = (const uint4*)&Bs[cb][(t_loc * 16 + lr) * 32 + hi * 16];
      bf[tn].q[0] = p[0];
      bf[tn].q[1] = p[1];
    }
    for (int tm = 0; tm < 2; ++tm)
      for (int tn = 0; tn < 4; ++tn)
        acc[tm][tn].v = wmma_bf16(af[tm].v, bf[tn].v, acc[tm][tn].v);
    __syncthreads();   // all reads of buffer cb done before it is re-filled
  }

  for (int tm = 0; tm < 2; ++tm)
    for (int tn = 0; tn < 4; ++tn)
      for (int r = 0; r < 8; ++r) {
        int row = rowBase + wm * 32 + tm * 16 + hi * 8 + r;
        int col = colBase + wn * 64 + tn * 16 + lr;
        C[(size_t)row * Ndim + col] = acc[tm][tn].e[r];
      }
}

// ---------------------------------------------------------------------------
// WMMA flash attention.  Q tile = 16 rows, key tiles of 32, HD = 128.
// V is pre-transposed per head so P@V fragments are contiguous global b128.
// grid (L/16, NH), block = 1 wave (32 threads).
// ---------------------------------------------------------------------------
__global__ __launch_bounds__(32)
void flash_attn_kernel(const __bf16* __restrict__ Qb,   // [L,NH,128]
                       const __bf16* __restrict__ Kb,   // [L,NH,128]
                       const __bf16* __restrict__ Vt,   // [NH*128][L]
                       const unsigned char* __restrict__ mask,
                       __bf16* __restrict__ Ob, int L, int NH) {
  const int HD = 128;
  int h  = blockIdx.y;
  int q0 = blockIdx.x * 16;
  int lane = threadIdx.x;
  int lr = lane & 15;
  int hi = lane >> 4;

  __shared__ __bf16 Ps[16][32];

  FragBF qf[4];
  {
    const __bf16* qp = Qb + ((size_t)(q0 + lr) * NH + h) * HD;
    for (int j = 0; j < 4; ++j) {
      qf[j].q[0] = *(const uint4*)(qp + j * 32 + hi * 16);
      qf[j].q[1] = *(const uint4*)(qp + j * 32 + hi * 16 + 8);
    }
  }

  float m[8], l[8];
  AccF oacc[8];
  for (int r = 0; r < 8; ++r) { m[r] = -1e30f; l[r] = 0.f; }
  for (int g = 0; g < 8; ++g)
    for (int r = 0; r < 8; ++r) oacc[g].e[r] = 0.f;

  const float scale = 0.08838834764831845f;   // 1/sqrt(128)
  int ntiles = L / 32;
  for (int kt = 0; kt < ntiles; ++kt) {
    int key0 = kt * 32;

    AccF sa, sb;
    for (int r = 0; r < 8; ++r) { sa.e[r] = 0.f; sb.e[r] = 0.f; }
    for (int j = 0; j < 4; ++j) {
      FragBF kfa, kfb;
      const __bf16* kpa = Kb + ((size_t)(key0 + lr) * NH + h) * HD + j * 32 + hi * 16;
      kfa.q[0] = *(const uint4*)(kpa);
      kfa.q[1] = *(const uint4*)(kpa + 8);
      const __bf16* kpb = Kb + ((size_t)(key0 + 16 + lr) * NH + h) * HD + j * 32 + hi * 16;
      kfb.q[0] = *(const uint4*)(kpb);
      kfb.q[1] = *(const uint4*)(kpb + 8);
      sa.v = wmma_bf16(qf[j].v, kfa.v, sa.v);
      sb.v = wmma_bf16(qf[j].v, kfb.v, sb.v);
    }

    bool ma = mask[key0 + lr] != 0;
    bool mb = mask[key0 + 16 + lr] != 0;
    float rs[8];
    for (int r = 0; r < 8; ++r) {
      float va = ma ? sa.e[r] * scale : -1e30f;
      float vb = mb ? sb.e[r] * scale : -1e30f;
      float tmax = fmaxf(va, vb);
      for (int off = 1; off < 16; off <<= 1)
        tmax = fmaxf(tmax, __shfl_xor(tmax, off, 16));
      float mnew = fmaxf(m[r], tmax);
      float sc = expf(m[r] - mnew);
      float pa = expf(va - mnew);
      float pb = expf(vb - mnew);
      float rowsum = pa + pb;
      for (int off = 1; off < 16; off <<= 1)
        rowsum += __shfl_xor(rowsum, off, 16);
      l[r] = l[r] * sc + rowsum;
      m[r] = mnew;
      rs[r] = sc;
      sa.e[r] = pa; sb.e[r] = pb;
    }
    for (int g = 0; g < 8; ++g)
      for (int r = 0; r < 8; ++r) oacc[g].e[r] *= rs[r];

    __syncthreads();
    for (int r = 0; r < 8; ++r) {
      int row = hi ? (r + 8) : r;
      Ps[row][lr]      = f2bf(sa.e[r]);
      Ps[row][16 + lr] = f2bf(sb.e[r]);
    }
    __syncthreads();
    FragBF pf;
    pf.q[0] = *(const uint4*)&Ps[lr][hi * 16];
    pf.q[1] = *(const uint4*)&Ps[lr][hi * 16 + 8];

    for (int g = 0; g < 8; ++g) {
      FragBF vf;
      const __bf16* vp = Vt + ((size_t)h * HD + g * 16 + lr) * L + key0 + hi * 16;
      vf.q[0] = *(const uint4*)(vp);
      vf.q[1] = *(const uint4*)(vp + 8);
      oacc[g].v = wmma_bf16(pf.v, vf.v, oacc[g].v);
    }
    __syncthreads();
  }

  for (int g = 0; g < 8; ++g)
    for (int r = 0; r < 8; ++r) {
      int row = hi ? (r + 8) : r;
      float o = oacc[g].e[r] / l[r];
      Ob[((size_t)(q0 + row) * NH + h) * HD + g * 16 + lr] = f2bf(o);
    }
}

// ---------------------------------------------------------------------------
// Host orchestration
// ---------------------------------------------------------------------------
extern "C" void kernel_launch(void* const* d_in, const int* in_sizes, int n_in,
                              void* d_out, int out_size, void* d_ws, size_t ws_size,
                              hipStream_t stream) {
  (void)in_sizes; (void)n_in; (void)out_size; (void)ws_size;
  const int LI = 2048, LT = 256, L = 2304, D = 2048, NH = 16, MM = 8192;
  const int SIXD = 6 * D, QKVD = 3 * D;

  const float* img        = (const float*)d_in[0];
  const float* txt        = (const float*)d_in[1];
  const float* vec        = (const float*)d_in[2];
  const float* fcos       = (const float*)d_in[3];
  const float* fsin       = (const float*)d_in[4];
  const unsigned char* mask = (const unsigned char*)d_in[5];
  const float* img_mod_w  = (const float*)d_in[6];
  const float* img_mod_b  = (const float*)d_in[7];
  const float* img_qkv_w  = (const float*)d_in[8];
  const float* img_qn_w   = (const float*)d_in[9];
  const float* img_kn_w   = (const float*)d_in[10];
  const float* img_proj_w = (const float*)d_in[11];
  const float* img_fc1_w  = (const float*)d_in[12];
  const float* img_fc1_b  = (const float*)d_in[13];
  const float* img_fc2_w  = (const float*)d_in[14];
  const float* img_fc2_b  = (const float*)d_in[15];
  const float* txt_mod_w  = (const float*)d_in[16];
  const float* txt_mod_b  = (const float*)d_in[17];
  const float* txt_qkv_w  = (const float*)d_in[18];
  const float* txt_qn_w   = (const float*)d_in[19];
  const float* txt_kn_w   = (const float*)d_in[20];
  const float* txt_proj_w = (const float*)d_in[21];
  const float* txt_fc1_w  = (const float*)d_in[22];
  const float* txt_fc1_b  = (const float*)d_in[23];
  const float* txt_fc2_w  = (const float*)d_in[24];
  const float* txt_fc2_b  = (const float*)d_in[25];

  float* out_img = (float*)d_out;
  float* out_txt = out_img + (size_t)LI * D;

  size_t cur = 0;
  auto alloc = [&](size_t bytes) -> char* {
    size_t off = (cur + 255) & ~(size_t)255;
    cur = off + bytes;
    return (char*)d_ws + off;
  };
  float*  vs      = (float*)alloc((size_t)D * 4);
  float*  im      = (float*)alloc((size_t)SIXD * 4);
  float*  tm      = (float*)alloc((size_t)SIXD * 4);
  __bf16* ixb     = (__bf16*)alloc((size_t)LI * D * 2);
  __bf16* txb     = (__bf16*)alloc((size_t)LT * D * 2);
  __bf16* wconv   = (__bf16*)alloc((size_t)D * MM * 2);       // packed-B arena
  float*  gemmout = (float*)alloc((size_t)LI * MM * 4);       // GEMM C arena
  __bf16* Qb      = (__bf16*)alloc((size_t)L * D * 2);
  __bf16* Kb      = (__bf16*)alloc((size_t)L * D * 2);
  __bf16* Vt      = (__bf16*)alloc((size_t)L * D * 2);
  __bf16* attnb   = (__bf16*)alloc((size_t)L * D * 2);
  __bf16* hb      = (__bf16*)alloc((size_t)LI * MM * 2);

  dim3 b256(256);
  auto eb = [](int n) { return dim3((n + 255) / 256); };

  copy_f32_kernel<<<eb(LI * D), b256, 0, stream>>>(img, out_img, LI * D);
  copy_f32_kernel<<<eb(LT * D), b256, 0, stream>>>(txt, out_txt, LT * D);

  silu_kernel<<<eb(D), b256, 0, stream>>>(vec, vs, D);
  modvec_kernel<<<eb(SIXD), b256, 0, stream>>>(vs, img_mod_w, img_mod_b, im, D, SIXD);
  modvec_kernel<<<eb(SIXD), b256, 0, stream>>>(vs, txt_mod_w, txt_mod_b, tm, D, SIXD);

  ln_modulate_cast_kernel<<<LI, b256, 0, stream>>>(out_img, im, im + D, ixb, D);
  ln_modulate_cast_kernel<<<LT, b256, 0, stream>>>(out_txt, tm, tm + D, txb, D);

  pack_b_kernel<<<eb(D * QKVD), b256, 0, stream>>>(img_qkv_w, wconv, D, QKVD);
  gemm_bf16_kernel<<<dim3(QKVD / 128, LI / 128), b256, 0, stream>>>(ixb, wconv, gemmout, LI, QKVD, D);
  pack_b_kernel<<<eb(D * QKVD), b256, 0, stream>>>(txt_qkv_w, wconv, D, QKVD);
  gemm_bf16_kernel<<<dim3(QKVD / 128, LT / 128), b256, 0, stream>>>(
      txb, wconv, gemmout + (size_t)LI * QKVD, LT, QKVD, D);

  qkv_post_kernel<<<dim3(LI, NH), dim3(128), 0, stream>>>(
      gemmout, img_qn_w, img_kn_w, fcos, fsin, 1, Qb, Kb, Vt, NH, L, 0);
  qkv_post_kernel<<<dim3(LT, NH), dim3(128), 0, stream>>>(
      gemmout + (size_t)LI * QKVD, txt_qn_w, txt_kn_w, fcos, fsin, 0, Qb, Kb, Vt, NH, L, LI);

  flash_attn_kernel<<<dim3(L / 16, NH), dim3(32), 0, stream>>>(Qb, Kb, Vt, mask, attnb, L, NH);

  pack_b_kernel<<<eb(D * D), b256, 0, stream>>>(img_proj_w, wconv, D, D);
  gemm_bf16_kernel<<<dim3(D / 128, LI / 128), b256, 0, stream>>>(attnb, wconv, gemmout, LI, D, D);
  gate_residual_kernel<<<eb(LI * D), b256, 0, stream>>>(out_img, gemmout, im + 2 * D, LI, D);

  pack_b_kernel<<<eb(D * D), b256, 0, stream>>>(txt_proj_w, wconv, D, D);
  gemm_bf16_kernel<<<dim3(D / 128, LT / 128), b256, 0, stream>>>(
      attnb + (size_t)LI * D, wconv, gemmout, LT, D, D);
  gate_residual_kernel<<<eb(LT * D), b256, 0, stream>>>(out_txt, gemmout, tm + 2 * D, LT, D);

  // ---- img MLP ----
  ln_modulate_cast_kernel<<<LI, b256, 0, stream>>>(out_img, im + 3 * D, im + 4 * D, ixb, D);
  pack_b_kernel<<<eb(D * MM), b256, 0, stream>>>(img_fc1_w, wconv, D, MM);
  gemm_bf16_kernel<<<dim3(MM / 128, LI / 128), b256, 0, stream>>>(ixb, wconv, gemmout, LI, MM, D);
  bias_gelu_cast_kernel<<<eb(LI * MM), b256, 0, stream>>>(gemmout, img_fc1_b, hb, LI * MM, MM);
  pack_b_kernel<<<eb(MM * D), b256, 0, stream>>>(img_fc2_w, wconv, MM, D);
  gemm_bf16_kernel<<<dim3(D / 128, LI / 128), b256, 0, stream>>>(hb, wconv, gemmout, LI, D, MM);
  bias_gate_residual_kernel<<<eb(LI * D), b256, 0, stream>>>(
      out_img, gemmout, img_fc2_b, im + 5 * D, LI, D);

  // ---- txt MLP ----
  ln_modulate_cast_kernel<<<LT, b256, 0, stream>>>(out_txt, tm + 3 * D, tm + 4 * D, txb, D);
  pack_b_kernel<<<eb(D * MM), b256, 0, stream>>>(txt_fc1_w, wconv, D, MM);
  gemm_bf16_kernel<<<dim3(MM / 128, LT / 128), b256, 0, stream>>>(txb, wconv, gemmout, LT, MM, D);
  bias_gelu_cast_kernel<<<eb(LT * MM), b256, 0, stream>>>(gemmout, txt_fc1_b, hb, LT * MM, MM);
  pack_b_kernel<<<eb(MM * D), b256, 0, stream>>>(txt_fc2_w, wconv, MM, D);
  gemm_bf16_kernel<<<dim3(D / 128, LT / 128), b256, 0, stream>>>(hb, wconv, gemmout, LT, D, MM);
  bias_gate_residual_kernel<<<eb(LT * D), b256, 0, stream>>>(
      out_txt, gemmout, txt_fc2_b, tm + 5 * D, LT, D);
}
